// SDSABlock_91233695301649
// MI455X (gfx1250) — 16-bit and 8-bit matrix operand order repaired and checked by reading, compiled, not run on hardware
//
#include <hip/hip_runtime.h>
#include <hip/hip_bf16.h>
#include <stdint.h>

typedef __attribute__((ext_vector_type(16))) __bf16 v16bf;
typedef __attribute__((ext_vector_type(8)))  float  v8f;
typedef __attribute__((ext_vector_type(8)))  int    v8i;

struct __align__(16) U4 { unsigned x, y, z, w; };
struct __align__(8)  U2 { unsigned x, y; };

union BfFrag { U4 u[2]; v16bf v; };
union I8Frag { U2 p[4]; U4 q[2]; v8i v; };

namespace {
constexpr int T_ = 4, B_ = 32, N_ = 197, D_ = 512, DH_ = 2048, H_ = 8, HD_ = 64;
constexpr int NP_ = 208;
constexpr int KP_ = 256;
constexpr int NT_ = NP_ / 16;
constexpr int DT_ = HD_ / 16;
constexpr int M_  = T_ * B_ * N_;
constexpr long long TS_  = (long long)B_ * N_ * D_;
constexpr long long TSH_ = (long long)B_ * N_ * DH_;

constexpr size_t SZ_W_D = (size_t)D_ * D_ * 2;
constexpr size_t SZ_W_H = (size_t)DH_ * D_ * 2;
constexpr size_t SZ_ABF = (size_t)M_ * D_ * 2;
constexpr size_t SZ_C   = (size_t)M_ * D_ * 4;
constexpr size_t SZ_QS  = (size_t)T_ * B_ * H_ * NP_ * HD_;
constexpr size_t SZ_VT  = (size_t)T_ * B_ * H_ * HD_ * KP_;
constexpr size_t SZ_H   = (size_t)M_ * DH_ * 2;

constexpr size_t OFF_WQ  = 0;
constexpr size_t OFF_WK  = OFF_WQ  + SZ_W_D;
constexpr size_t OFF_WV  = OFF_WK  + SZ_W_D;
constexpr size_t OFF_WP  = OFF_WV  + SZ_W_D;
constexpr size_t OFF_W1  = OFF_WP  + SZ_W_D;
constexpr size_t OFF_W2  = OFF_W1  + SZ_W_H;
constexpr size_t OFF_ABF = OFF_W2  + SZ_W_H;
constexpr size_t OFF_CQ  = OFF_ABF + SZ_ABF;
constexpr size_t OFF_CK  = OFF_CQ  + SZ_C;
constexpr size_t OFF_CV  = OFF_CK  + SZ_C;
constexpr size_t OFF_QS  = OFF_CV  + SZ_C;
constexpr size_t OFF_KS  = OFF_QS  + SZ_QS;
constexpr size_t OFF_VT  = OFF_KS  + SZ_QS;
constexpr size_t OFF_H   = OFF_VT  + SZ_VT;
}

__device__ __forceinline__ unsigned lds_off(const void* p) {
    return (unsigned)(uintptr_t)p;
}
__device__ __forceinline__ void async_cp16(unsigned lds, const void* gptr) {
    asm volatile("global_load_async_to_lds_b128 %0, %1, off"
                 :: "v"(lds), "v"(gptr) : "memory");
}
__device__ __forceinline__ void wait_async_le6() {
    asm volatile("s_wait_asynccnt 0x6" ::: "memory");
}
__device__ __forceinline__ void wait_async_0() {
    asm volatile("s_wait_asynccnt 0x0" ::: "memory");
}

__global__ void cast_bf16_kernel(const float* __restrict__ in,
                                 __hip_bfloat16* __restrict__ out, int n) {
    int i = blockIdx.x * 256 + threadIdx.x;
    if (i < n) out[i] = __float2bfloat16(in[i]);
}

__global__ void zero_u32_kernel(unsigned* __restrict__ p, long long n) {
    long long i = (long long)blockIdx.x * 256 + threadIdx.x;
    if (i < n) p[i] = 0u;
}

__global__ __launch_bounds__(256)
void ln_kernel(const float* __restrict__ X, const float* __restrict__ g,
               const float* __restrict__ b, __hip_bfloat16* __restrict__ Y) {
    __shared__ float rs[256];
    __shared__ float rq[256];
    const int row = blockIdx.x, tid = threadIdx.x;
    const float* xr = X + (size_t)row * D_;
    const float a = xr[tid], c = xr[tid + 256];
    rs[tid] = a + c;
    rq[tid] = a * a + c * c;
    __syncthreads();
    for (int o = 128; o > 0; o >>= 1) {
        if (tid < o) { rs[tid] += rs[tid + o]; rq[tid] += rq[tid + o]; }
        __syncthreads();
    }
    const float mean = rs[0] * (1.0f / D_);
    const float var  = rq[0] * (1.0f / D_) - mean * mean;
    const float inv  = rsqrtf(var + 1e-5f);
    __hip_bfloat16* yr = Y + (size_t)row * D_;
    yr[tid]       = __float2bfloat16((a - mean) * inv * g[tid] + b[tid]);
    yr[tid + 256] = __float2bfloat16((c - mean) * inv * g[tid + 256] + b[tid + 256]);
}

template <bool OUT_BF16>
__global__ __launch_bounds__(256)
void gemm_kernel(const __bf16* __restrict__ A, const __bf16* __restrict__ W,
                 const float* __restrict__ bias, void* __restrict__ Cout,
                 const int K, const int Nout) {
    __shared__ __align__(16) unsigned char SA[2][128 * 128];
    __shared__ __align__(16) unsigned char SB[2][64 * 128];

    const int tid  = threadIdx.x;
    const int lane = tid & 31;
    const int wid  = tid >> 5;
    const int lr = lane & 15;
    const int lh = lane >> 4;
    const int mblock = blockIdx.x * 128;
    const int nblock = blockIdx.y * 64;
    const int mbase = (wid & 3) * 32;
    const int nbase = (wid >> 2) * 32;

    const int srow = tid >> 3;
    const int skc  = tid & 7;
    const __bf16* gA = A + (size_t)(mblock + srow) * K + skc * 8;
    const __bf16* gB = W + (size_t)(nblock + srow) * K + skc * 8;
    const size_t aStep = (size_t)32 * K;
    const unsigned lA0 = lds_off(&SA[0][srow * 128 + skc * 16]);
    const unsigned lB0 = lds_off(&SB[0][srow * 128 + skc * 16]);

    v8f acc[2][2];
#pragma unroll
    for (int i = 0; i < 2; ++i)
#pragma unroll
        for (int j = 0; j < 2; ++j)
#pragma unroll
            for (int e = 0; e < 8; ++e) acc[i][j][e] = 0.0f;

    const int stages = K >> 6;
    async_cp16(lA0,         gA);
    async_cp16(lA0 + 4096,  gA + aStep);
    async_cp16(lA0 + 8192,  gA + 2 * aStep);
    async_cp16(lA0 + 12288, gA + 3 * aStep);
    async_cp16(lB0,         gB);
    async_cp16(lB0 + 4096,  gB + aStep);

    for (int s = 0; s < stages; ++s) {
        const int cur = s & 1;
        if (s + 1 < stages) {
            const int k0 = (s + 1) << 6;
            const unsigned lA = lA0 + (cur ^ 1) * (128 * 128);
            const unsigned lB = lB0 + (cur ^ 1) * (64 * 128);
            async_cp16(lA,         gA + k0);
            async_cp16(lA + 4096,  gA + k0 + aStep);
            async_cp16(lA + 8192,  gA + k0 + 2 * aStep);
            async_cp16(lA + 12288, gA + k0 + 3 * aStep);
            async_cp16(lB,         gB + k0);
            async_cp16(lB + 4096,  gB + k0 + aStep);
            wait_async_le6();
        } else {
            wait_async_0();
        }
        __syncthreads();

#pragma unroll
        for (int ks = 0; ks < 2; ++ks) {
            BfFrag a0, a1, b0, b1;
            const unsigned char* pa = &SA[cur][(mbase + lr) * 128 + ks * 64 + lh * 16];
            a0.u[0] = *(const U4*)(pa);
            a0.u[1] = *(const U4*)(pa + 32);
            a1.u[0] = *(const U4*)(pa + 2048);
            a1.u[1] = *(const U4*)(pa + 2048 + 32);
            const unsigned char* pb = &SB[cur][(nbase + lr) * 128 + ks * 64 + lh * 16];
            b0.u[0] = *(const U4*)(pb);
            b0.u[1] = *(const U4*)(pb + 32);
            b1.u[0] = *(const U4*)(pb + 2048);
            b1.u[1] = *(const U4*)(pb + 2048 + 32);
            acc[0][0] = __builtin_amdgcn_wmma_f32_16x16x32_bf16(false, a0.v, false, b0.v, (short)0, acc[0][0], false, false);
            acc[0][1] = __builtin_amdgcn_wmma_f32_16x16x32_bf16(false, a0.v, false, b1.v, (short)0, acc[0][1], false, false);
            acc[1][0] = __builtin_amdgcn_wmma_f32_16x16x32_bf16(false, a1.v, false, b0.v, (short)0, acc[1][0], false, false);
            acc[1][1] = __builtin_amdgcn_wmma_f32_16x16x32_bf16(false, a1.v, false, b1.v, (short)0, acc[1][1], false, false);
        }

        __syncthreads();
    }

#pragma unroll
    for (int si = 0; si < 2; ++si)
#pragma unroll
        for (int sj = 0; sj < 2; ++sj) {
            const int col = nblock + nbase + sj * 16 + lr;
            const float bv = bias ? bias[col] : 0.0f;
            const int m0 = mblock + mbase + si * 16 + lh * 8;
#pragma unroll
            for (int r = 0; r < 8; ++r) {
                const float val = acc[si][sj][r] + bv;
                const size_t idx = (size_t)(m0 + r) * Nout + col;
                if (OUT_BF16) ((__hip_bfloat16*)Cout)[idx] = __float2bfloat16(val);
                else          ((float*)Cout)[idx] = val;
            }
        }
}

__global__ void lif_qkv_kernel(const float* __restrict__ cq, const float* __restrict__ ck,
                               const float* __restrict__ cv, unsigned char* __restrict__ QS,
                               unsigned char* __restrict__ KS, unsigned char* __restrict__ VT) {
    const long long j = (long long)blockIdx.x * 256 + threadIdx.x;
    if (j >= TS_) return;
    const int d  = (int)(j & (D_ - 1));
    const int n  = (int)((j >> 9) % N_);
    const int bb = (int)(j / ((long long)N_ * D_));
    const int h = d >> 6, dh = d & 63;
    float vq = 0.0f, vk = 0.0f, vv = 0.0f;
#pragma unroll
    for (int t = 0; t < T_; ++t) {
        const long long off = t * TS_ + j;
        vq = 0.5f * (vq + cq[off]);
        vk = 0.5f * (vk + ck[off]);
        vv = 0.5f * (vv + cv[off]);
        const unsigned char sq = vq >= 1.0f, sk = vk >= 1.0f, sv = vv >= 1.0f;
        if (sq) vq = 0.0f;
        if (sk) vk = 0.0f;
        if (sv) vv = 0.0f;
        const int tbh = (t * B_ + bb) * H_ + h;
        QS[(size_t)tbh * NP_ * HD_ + (size_t)n * HD_ + dh] = sq;
        KS[(size_t)tbh * NP_ * HD_ + (size_t)n * HD_ + dh] = sk;
        VT[(size_t)tbh * HD_ * KP_ + (size_t)dh * KP_ + n] = sv;
    }
}

__global__ __launch_bounds__(256)
void attn_kernel(const unsigned char* __restrict__ QS, const unsigned char* __restrict__ KS,
                 const unsigned char* __restrict__ VT, __hip_bfloat16* __restrict__ O) {
    __shared__ __align__(16) unsigned char S[NP_ * KP_];
    const int tbh = blockIdx.x;
    const int h  = tbh & (H_ - 1);
    const int tb = tbh >> 3;
    const int lane = threadIdx.x & 31;
    const int wid  = threadIdx.x >> 5;
    const int lr = lane & 15, lh = lane >> 4;

    for (int i = threadIdx.x; i < NP_ * KP_ / 4; i += 256) ((unsigned*)S)[i] = 0u;
    __syncthreads();

    const unsigned char* qb = QS + (size_t)tbh * NP_ * HD_;
    const unsigned char* kb = KS + (size_t)tbh * NP_ * HD_;
    const unsigned char* vb = VT + (size_t)tbh * HD_ * KP_;

    for (int job = wid; job < NT_ * NT_; job += 8) {
        const int ti = job / NT_, tj = job % NT_;
        I8Frag a, b;
        const unsigned char* ap = qb + (size_t)(ti * 16 + lr) * HD_ + lh * 16;
        a.p[0] = *(const U2*)(ap);
        a.p[1] = *(const U2*)(ap + 8);
        a.p[2] = *(const U2*)(ap + 32);
        a.p[3] = *(const U2*)(ap + 40);
        const unsigned char* bp = kb + (size_t)(tj * 16 + lr) * HD_ + lh * 16;
        b.q[0] = *(const U4*)(bp);
        b.q[1] = *(const U4*)(bp + 32);
        v8i c;
#pragma unroll
        for (int e = 0; e < 8; ++e) c[e] = 0;
        c = __builtin_amdgcn_wmma_i32_16x16x64_iu8(false, a.v, false, b.v, c, false, false);
        unsigned char* srow = &S[(size_t)(ti * 16 + lh * 8) * KP_ + tj * 16 + lr];
#pragma unroll
        for (int r = 0; r < 8; ++r) srow[(size_t)r * KP_] = (unsigned char)c[r];
    }
    __syncthreads();

    for (int job = wid; job < NT_ * DT_; job += 8) {
        const int ti = job >> 2, td = job & 3;
        v8i acc;
#pragma unroll
        for (int e = 0; e < 8; ++e) acc[e] = 0;
        const unsigned char* arow = &S[(size_t)(ti * 16 + lr) * KP_ + lh * 16];
        const unsigned char* brow = vb + (size_t)(td * 16 + lr) * KP_ + lh * 16;
#pragma unroll
        for (int k0 = 0; k0 < KP_; k0 += 64) {
            I8Frag a, b;
            a.p[0] = *(const U2*)(arow + k0);
            a.p[1] = *(const U2*)(arow + k0 + 8);
            a.p[2] = *(const U2*)(arow + k0 + 32);
            a.p[3] = *(const U2*)(arow + k0 + 40);
            b.q[0] = *(const U4*)(brow + k0);
            b.q[1] = *(const U4*)(brow + k0 + 32);
            acc = __builtin_amdgcn_wmma_i32_16x16x64_iu8(false, a.v, false, b.v, acc, false, false);
        }
        const int col = h * HD_ + td * 16 + lr;
        const int i0 = ti * 16 + lh * 8;
#pragma unroll
        for (int r = 0; r < 8; ++r) {
            const int i = i0 + r;
            if (i < N_) {
                const size_t m = (size_t)tb * N_ + i;
                O[m * D_ + col] = __float2bfloat16((float)acc[r] * 0.125f);
            }
        }
    }
}

__global__ void lif_add_kernel(const float* __restrict__ cur, const float* __restrict__ res,
                               float* __restrict__ out) {
    const long long j = (long long)blockIdx.x * 256 + threadIdx.x;
    if (j >= TS_) return;
    float v = 0.0f;
#pragma unroll
    for (int t = 0; t < T_; ++t) {
        const long long off = t * TS_ + j;
        v = 0.5f * (v + cur[off]);
        const float s = (v >= 1.0f) ? 1.0f : 0.0f;
        out[off] = res[off] + s;
        v *= (1.0f - s);
    }
}

__global__ void lif_inplace_bf16_kernel(__hip_bfloat16* __restrict__ buf) {
    const long long j = (long long)blockIdx.x * 256 + threadIdx.x;
    if (j >= TSH_) return;
    float v = 0.0f;
#pragma unroll
    for (int t = 0; t < T_; ++t) {
        const long long off = t * TSH_ + j;
        v = 0.5f * (v + __bfloat162float(buf[off]));
        const float s = (v >= 1.0f) ? 1.0f : 0.0f;
        buf[off] = __float2bfloat16(s);
        v *= (1.0f - s);
    }
}

extern "C" void kernel_launch(void* const* d_in, const int* in_sizes, int n_in,
                              void* d_out, int out_size, void* d_ws, size_t ws_size,
                              hipStream_t stream) {
    (void)in_sizes; (void)n_in; (void)out_size; (void)ws_size;
    const float* x     = (const float*)d_in[0];
    const float* ln1_g = (const float*)d_in[1];
    const float* ln1_b = (const float*)d_in[2];
    const float* wq    = (const float*)d_in[3];
    const float* wk    = (const float*)d_in[4];
    const float* wv    = (const float*)d_in[5];
    const float* pw    = (const float*)d_in[6];
    const float* pb    = (const float*)d_in[7];
    const float* ln2_g = (const float*)d_in[8];
    const float* ln2_b = (const float*)d_in[9];
    const float* w1    = (const float*)d_in[10];
    const float* b1    = (const float*)d_in[11];
    const float* w2    = (const float*)d_in[12];
    const float* b2    = (const float*)d_in[13];

    char* ws = (char*)d_ws;
    __hip_bfloat16* WQ  = (__hip_bfloat16*)(ws + OFF_WQ);
    __hip_bfloat16* WK  = (__hip_bfloat16*)(ws + OFF_WK);
    __hip_bfloat16* WV  = (__hip_bfloat16*)(ws + OFF_WV);
    __hip_bfloat16* WP  = (__hip_bfloat16*)(ws + OFF_WP);
    __hip_bfloat16* W1  = (__hip_bfloat16*)(ws + OFF_W1);
    __hip_bfloat16* W2  = (__hip_bfloat16*)(ws + OFF_W2);
    __hip_bfloat16* ABF = (__hip_bfloat16*)(ws + OFF_ABF);
    float* CQ = (float*)(ws + OFF_CQ);
    float* CK = (float*)(ws + OFF_CK);
    float* CV = (float*)(ws + OFF_CV);
    unsigned char* QS = (unsigned char*)(ws + OFF_QS);
    unsigned char* KS = (unsigned char*)(ws + OFF_KS);
    unsigned char* VT = (unsigned char*)(ws + OFF_VT);
    __hip_bfloat16* Hb = (__hip_bfloat16*)(ws + OFF_H);

    const unsigned gE  = (unsigned)((TS_ + 255) / 256);
    const unsigned gEH = (unsigned)((TSH_ + 255) / 256);
    const dim3 g512(M_ / 128, D_ / 64);
    const dim3 g2048(M_ / 128, DH_ / 64);

    cast_bf16_kernel<<<(D_ * D_) / 256, 256, 0, stream>>>(wq, WQ, D_ * D_);
    cast_bf16_kernel<<<(D_ * D_) / 256, 256, 0, stream>>>(wk, WK, D_ * D_);
    cast_bf16_kernel<<<(D_ * D_) / 256, 256, 0, stream>>>(wv, WV, D_ * D_);
    cast_bf16_kernel<<<(D_ * D_) / 256, 256, 0, stream>>>(pw, WP, D_ * D_);
    cast_bf16_kernel<<<(DH_ * D_) / 256, 256, 0, stream>>>(w1, W1, DH_ * D_);
    cast_bf16_kernel<<<(DH_ * D_) / 256, 256, 0, stream>>>(w2, W2, DH_ * D_);
    zero_u32_kernel<<<(unsigned)((2 * SZ_QS + SZ_VT) / 4 / 256), 256, 0, stream>>>(
        (unsigned*)QS, (long long)((2 * SZ_QS + SZ_VT) / 4));

    ln_kernel<<<M_, 256, 0, stream>>>(x, ln1_g, ln1_b, ABF);
    gemm_kernel<false><<<g512, 256, 0, stream>>>((const __bf16*)ABF, (const __bf16*)WQ, nullptr, CQ, D_, D_);
    gemm_kernel<false><<<g512, 256, 0, stream>>>((const __bf16*)ABF, (const __bf16*)WK, nullptr, CK, D_, D_);
    gemm_kernel<false><<<g512, 256, 0, stream>>>((const __bf16*)ABF, (const __bf16*)WV, nullptr, CV, D_, D_);
    lif_qkv_kernel<<<gE, 256, 0, stream>>>(CQ, CK, CV, QS, KS, VT);
    attn_kernel<<<T_ * B_ * H_, 256, 0, stream>>>(QS, KS, VT, ABF);
    gemm_kernel<false><<<g512, 256, 0, stream>>>((const __bf16*)ABF, (const __bf16*)WP, pb, CQ, D_, D_);
    lif_add_kernel<<<gE, 256, 0, stream>>>(CQ, x, CK);

    ln_kernel<<<M_, 256, 0, stream>>>(CK, ln2_g, ln2_b, ABF);
    gemm_kernel<true><<<g2048, 256, 0, stream>>>((const __bf16*)ABF, (const __bf16*)W1, b1, Hb, D_, DH_);
    lif_inplace_bf16_kernel<<<gEH, 256, 0, stream>>>(Hb);
    gemm_kernel<false><<<g512, 256, 0, stream>>>((const __bf16*)Hb, (const __bf16*)W2, b2, CV, DH_, D_);
    lif_add_kernel<<<gE, 256, 0, stream>>>(CV, CK, (float*)d_out);
}
